// RotaryTransformerLayer_68796786147576
// MI455X (gfx1250) — matrix operand order repaired and checked by reading, not compiled, not run on hardware
//
#include <hip/hip_runtime.h>
#include <cstdint>
#include <cstddef>

// ---------------------------------------------------------------------------
// CDNA5 (gfx1250) transformer layer: all GEMMs + attention on
// v_wmma_f32_16x16x32_bf16 (wave32), f32 softmax / LN / RoPE epilogues.
// GEMM staging uses the gfx1250 async global->LDS path (ASYNCcnt-tracked).
// ---------------------------------------------------------------------------

typedef __bf16 bf16_t;
typedef __attribute__((ext_vector_type(16))) __bf16 v16bf;
typedef __attribute__((ext_vector_type(8)))  float  v8f;

union Frag16 {            // one 16-bit WMMA operand fragment (16 bf16 / lane)
    v16bf v;
    uint4 q[2];
};

#define WMMA_BF16(A, B, C) \
    __builtin_amdgcn_wmma_f32_16x16x32_bf16(false, (A), false, (B), (short)0, (C), false, false)

// ---- gfx1250 async global->LDS copy (ASYNCcnt-tracked), guarded -----------
#if defined(__has_builtin)
#if __has_builtin(__builtin_amdgcn_global_load_async_to_lds_b128)
#define HAVE_ASYNC_COPY 1
#else
#define HAVE_ASYNC_COPY 0
#endif
#else
#define HAVE_ASYNC_COPY 0
#endif

#if HAVE_ASYNC_COPY
// Builtin signature (from hipcc diagnostic): param 1 is
//   int __vector_size__(16) addrspace(1)* (global source),
// param 2 the LDS destination, then imm offset + imm cpol.
typedef int b128vec __attribute__((vector_size(16)));
typedef __attribute__((address_space(1))) b128vec* gptr_b128;
typedef __attribute__((address_space(3))) b128vec* lptr_b128;

#define ASYNC_CP16(ldst, gsrc)                                                  \
    __builtin_amdgcn_global_load_async_to_lds_b128(                             \
        (gptr_b128)(gsrc), (lptr_b128)(ldst), 0, 0)

__device__ __forceinline__ void async_wait_all() {
#if __has_builtin(__builtin_amdgcn_s_wait_asynccnt)
    __builtin_amdgcn_s_wait_asynccnt(0);
#else
    asm volatile("s_wait_asynccnt 0" ::: "memory");
#endif
}
#endif

static constexpr int D_MODEL = 1024;
static constexpr int NHEAD   = 16;
static constexpr int SEQ     = 2048;
static constexpr int BATCH   = 2;
static constexpr int BL      = BATCH * SEQ;     // 4096 rows

// ---------------------------------------------------------------------------
// f32 -> bf16 elementwise convert
// ---------------------------------------------------------------------------
__global__ __launch_bounds__(256)
void cvt_f32_bf16_kernel(const float* __restrict__ in, bf16_t* __restrict__ out, size_t n) {
    size_t i = (size_t)blockIdx.x * 256 + threadIdx.x;
    if (i < n) out[i] = (bf16_t)in[i];
}

// ---------------------------------------------------------------------------
// Transpose-convert: W[K][N] f32  ->  Wt[N][K] bf16   (LDS 32x32 tiles)
// ---------------------------------------------------------------------------
__global__ __launch_bounds__(256)
void transpose_bf16_kernel(const float* __restrict__ W, bf16_t* __restrict__ Wt,
                           int K, int N) {
    __shared__ float tile[32][33];
    const int tx = threadIdx.x & 31;
    const int ty = threadIdx.x >> 5;            // 0..7
    const int k0 = blockIdx.y * 32;
    const int n0 = blockIdx.x * 32;
#pragma unroll
    for (int i = 0; i < 4; ++i)
        tile[ty + i * 8][tx] = W[(size_t)(k0 + ty + i * 8) * N + n0 + tx];
    __syncthreads();
#pragma unroll
    for (int i = 0; i < 4; ++i)
        Wt[(size_t)(n0 + ty + i * 8) * K + k0 + tx] = (bf16_t)tile[tx][ty + i * 8];
}

// ---------------------------------------------------------------------------
// Generic bf16 WMMA GEMM:  C[M][N] = A[M][K] * Wt[N][K]^T + bias
//   EPI 0: outF = C
//   EPI 1: outF = C + resid                (pre-LN sum)
//   EPI 2: outB = bf16(gelu_exact(C))
//   EPI 3: outB = bf16(C)
// Workgroup tile 128x128, 8 waves (2M x 4N) of 64x32, K-step 64, LDS-staged
// via async global->LDS when available.  16 WMMAs per wave per barrier pair.
// ---------------------------------------------------------------------------
template <int EPI>
__global__ __launch_bounds__(256)
void gemm_bf16_kernel(const bf16_t* __restrict__ A, const bf16_t* __restrict__ Bt,
                      const float* __restrict__ bias, const float* __restrict__ resid,
                      float* __restrict__ outF, bf16_t* __restrict__ outB,
                      int M, int N, int K) {
    constexpr int TM = 128, TN = 128, TK = 64, LDT = TK + 8;  // 144B row stride
    __shared__ __align__(16) bf16_t lA[TM * LDT];
    __shared__ __align__(16) bf16_t lB[TN * LDT];

    const int tid  = threadIdx.x;
    const int wave = tid >> 5;
    const int lane = tid & 31;
    const int lh   = lane >> 4;                 // 0/1  (half-wave)
    const int ll   = lane & 15;
    const int m0   = blockIdx.y * TM;
    const int n0   = blockIdx.x * TN;
    const int wm   = (wave >> 2) * 64;          // wave M offset 0/64
    const int wn   = (wave & 3) * 32;           // wave N offset 0/32/64/96

    v8f acc[4][2] = {};                         // 4 M-subtiles x 2 N-subtiles

    for (int k0 = 0; k0 < K; k0 += TK) {
        // stage A/B tiles: 128x64 bf16 each = 1024 x 16B chunks, 4 per thread
        {
            int idx = tid;
#pragma unroll
            for (int rep = 0; rep < 4; ++rep, idx += 256) {
                int row = idx >> 3, c = idx & 7;
#if HAVE_ASYNC_COPY
                ASYNC_CP16(lA + row * LDT + c * 8,
                           A + (size_t)(m0 + row) * K + k0 + c * 8);
#else
                *(uint4*)(lA + row * LDT + c * 8) =
                    *(const uint4*)(A + (size_t)(m0 + row) * K + k0 + c * 8);
#endif
            }
            idx = tid;
#pragma unroll
            for (int rep = 0; rep < 4; ++rep, idx += 256) {
                int row = idx >> 3, c = idx & 7;
#if HAVE_ASYNC_COPY
                ASYNC_CP16(lB + row * LDT + c * 8,
                           Bt + (size_t)(n0 + row) * K + k0 + c * 8);
#else
                *(uint4*)(lB + row * LDT + c * 8) =
                    *(const uint4*)(Bt + (size_t)(n0 + row) * K + k0 + c * 8);
#endif
            }
#if HAVE_ASYNC_COPY
            async_wait_all();                   // s_wait_asynccnt 0
#endif
        }
        __syncthreads();

#pragma unroll
        for (int ks = 0; ks < 2; ++ks) {        // two 32-wide k-substeps
            Frag16 a[4], b[2];
#pragma unroll
            for (int i = 0; i < 4; ++i) {       // A frag: K = 16*lh + e (contig)
                const bf16_t* base = lA + (wm + i * 16 + ll) * LDT + ks * 32 + lh * 16;
                a[i].q[0] = *(const uint4*)(base);
                a[i].q[1] = *(const uint4*)(base + 8);
            }
#pragma unroll
            for (int j = 0; j < 2; ++j) {       // B frag: K = 16*lh + e (contig)
                const bf16_t* base = lB + (wn + j * 16 + ll) * LDT + ks * 32 + lh * 16;
                b[j].q[0] = *(const uint4*)(base);
                b[j].q[1] = *(const uint4*)(base + 8);
            }
#pragma unroll
            for (int i = 0; i < 4; ++i)
#pragma unroll
                for (int j = 0; j < 2; ++j)
                    acc[i][j] = WMMA_BF16(a[i].v, b[j].v, acc[i][j]);
        }
        __syncthreads();
    }

    // Epilogue.  D layout: lane holds col = ll, rows r + 8*lh (r = 0..7)
#pragma unroll
    for (int i = 0; i < 4; ++i) {
#pragma unroll
        for (int j = 0; j < 2; ++j) {
            const int col = n0 + wn + j * 16 + ll;
            const float bv = bias[col];
#pragma unroll
            for (int r = 0; r < 8; ++r) {
                const int row = m0 + wm + i * 16 + r + 8 * lh;
                float v = acc[i][j][r] + bv;
                const size_t o = (size_t)row * N + col;
                if (EPI == 1)      outF[o] = v + resid[o];
                else if (EPI == 2) { float g = 0.5f * v * (1.0f + erff(v * 0.70710678118f));
                                     outB[o] = (bf16_t)g; }
                else if (EPI == 3) outB[o] = (bf16_t)v;
                else               outF[o] = v;
            }
        }
    }
}

// ---------------------------------------------------------------------------
// RoPE + scatter:  qkv[BL][3072] bf16 ->
//   qh[bh][pos][64] (scaled by 1/sqrt(hd)), kh[bh][pos][64], vt[bh][64][pos]
// one thread per (b,pos,h,dpair)   dpair = 0..31
// ---------------------------------------------------------------------------
__global__ __launch_bounds__(256)
void rope_scatter_kernel(const bf16_t* __restrict__ qkv,
                         bf16_t* __restrict__ qh, bf16_t* __restrict__ kh,
                         bf16_t* __restrict__ vt) {
    const int idx = blockIdx.x * 256 + threadIdx.x;   // 2*2048*16*32 total
    const int d   = idx & 31;
    const int h   = (idx >> 5) & 15;
    const int pos = (idx >> 9) & 2047;
    const int b   = idx >> 20;

    const bf16_t* base = qkv + ((size_t)b * SEQ + pos) * 3072;
    const float q1 = (float)base[h * 64 + d],        q2 = (float)base[h * 64 + d + 32];
    const float k1 = (float)base[1024 + h * 64 + d], k2 = (float)base[1024 + h * 64 + d + 32];
    const float v1 = (float)base[2048 + h * 64 + d], v2 = (float)base[2048 + h * 64 + d + 32];

    const float inv_freq = __powf(10000.0f, -(float)(2 * d) / 64.0f);
    float s, c;
    __sincosf((float)pos * inv_freq, &s, &c);

    const int bh = b * NHEAD + h;
    const size_t qo = ((size_t)bh * SEQ + pos) * 64;
    const float qs = 0.125f;                         // 1/sqrt(64) folded into Q
    qh[qo + d]      = (bf16_t)((q1 * c - q2 * s) * qs);
    qh[qo + d + 32] = (bf16_t)((q2 * c + q1 * s) * qs);
    kh[qo + d]      = (bf16_t)(k1 * c - k2 * s);
    kh[qo + d + 32] = (bf16_t)(k2 * c + k1 * s);
    vt[((size_t)bh * 64 + d)      * SEQ + pos] = (bf16_t)v1;
    vt[((size_t)bh * 64 + d + 32) * SEQ + pos] = (bf16_t)v2;
}

// ---------------------------------------------------------------------------
// Flash attention:  per workgroup = (bh, 128 q-rows); per wave = 16 q-rows.
// 128-key blocks: S = Q K^T (16 WMMA), online f32 softmax, P V (16 WMMA,
// P re-fragmented through a padded per-wave LDS tile).  Next K/V blocks are
// prefetched (global_prefetch_b8), one cacheline per lane.
// ---------------------------------------------------------------------------
__global__ __launch_bounds__(256)
void attn_kernel(const bf16_t* __restrict__ qh, const bf16_t* __restrict__ kh,
                 const bf16_t* __restrict__ vt, bf16_t* __restrict__ attn) {
    constexpr int KB = 128;                    // key block
    constexpr int LDP = KB + 8;                // padded P row stride (272B)
    __shared__ __align__(16) bf16_t lP[8][16 * LDP];

    const int wave = threadIdx.x >> 5;
    const int lane = threadIdx.x & 31;
    const int lh   = lane >> 4, ll = lane & 15;
    const int bh   = blockIdx.y;                      // 0..31
    const int qbase = blockIdx.x * 128 + wave * 16;   // this wave's q rows

    // Q A-fragments (held in registers for the whole key sweep)
    Frag16 aq[2];
    const bf16_t* qrow = qh + ((size_t)bh * SEQ + qbase) * 64;
#pragma unroll
    for (int kk = 0; kk < 2; ++kk) {
        const bf16_t* base = qrow + (size_t)ll * 64 + kk * 32 + lh * 16;
        aq[kk].q[0] = *(const uint4*)(base);
        aq[kk].q[1] = *(const uint4*)(base + 8);
    }

    float m_i[8], l_i[8];
    v8f acc_o[4] = {};
#pragma unroll
    for (int r = 0; r < 8; ++r) { m_i[r] = -3.0e38f; l_i[r] = 0.0f; }

    const bf16_t* kbase = kh + (size_t)bh * SEQ * 64;
    const bf16_t* vbase = vt + (size_t)bh * 64 * SEQ;
    bf16_t* lp = &lP[wave][0];

    for (int kb = 0; kb < SEQ / KB; ++kb) {
        // prefetch next key/value blocks (one 256B line per lane)
        if (kb + 1 < SEQ / KB) {
            const bf16_t* nk = kbase + (size_t)(kb + 1) * KB * 64;
            __builtin_prefetch(nk + (size_t)lane * 128, 0, 3);
            __builtin_prefetch(nk + (size_t)(lane + 32) * 128, 0, 3);
            const bf16_t* nv = vbase + (size_t)(kb + 1) * KB;
            __builtin_prefetch(nv + (size_t)(2 * lane) * SEQ, 0, 3);
            __builtin_prefetch(nv + (size_t)(2 * lane + 1) * SEQ, 0, 3);
        }
        // ---- S = Q K^T  (16 x 128 block) ----
        v8f s[8] = {};
#pragma unroll
        for (int kk = 0; kk < 2; ++kk) {
#pragma unroll
            for (int j = 0; j < 8; ++j) {
                Frag16 bk;
                const bf16_t* base =
                    kbase + (size_t)(kb * KB + j * 16 + ll) * 64 + kk * 32 + lh * 16;
                bk.q[0] = *(const uint4*)(base);
                bk.q[1] = *(const uint4*)(base + 8);
                s[j] = WMMA_BF16(aq[kk].v, bk.v, s[j]);
            }
        }
        // ---- online softmax (per row r+8*lh, cols across 16-lane group) ----
#pragma unroll
        for (int r = 0; r < 8; ++r) {
            float mx = s[0][r];
#pragma unroll
            for (int j = 1; j < 8; ++j) mx = fmaxf(mx, s[j][r]);
#pragma unroll
            for (int msk = 8; msk >= 1; msk >>= 1) mx = fmaxf(mx, __shfl_xor(mx, msk, 16));
            const float mn = fmaxf(m_i[r], mx);
            const float alpha = __expf(m_i[r] - mn);
            float ps = 0.0f;
#pragma unroll
            for (int j = 0; j < 8; ++j) {
                float p = __expf(s[j][r] - mn);
                s[j][r] = p;
                ps += p;
            }
#pragma unroll
            for (int msk = 8; msk >= 1; msk >>= 1) ps += __shfl_xor(ps, msk, 16);
            l_i[r] = l_i[r] * alpha + ps;
            m_i[r] = mn;
#pragma unroll
            for (int j = 0; j < 4; ++j) acc_o[j][r] *= alpha;
        }
        // ---- P -> LDS (D layout scatter), then reload as A fragments ----
#pragma unroll
        for (int j = 0; j < 8; ++j)
#pragma unroll
            for (int r = 0; r < 8; ++r)
                lp[(r + 8 * lh) * LDP + j * 16 + ll] = (bf16_t)s[j][r];

#pragma unroll
        for (int kk = 0; kk < 4; ++kk) {        // 128 keys / 32 per WMMA step
            Frag16 ap;
            const bf16_t* base = lp + ll * LDP + kk * 32 + lh * 16;
            ap.q[0] = *(const uint4*)(base);
            ap.q[1] = *(const uint4*)(base + 8);
#pragma unroll
            for (int j = 0; j < 4; ++j) {
                Frag16 bv;
                const bf16_t* vb =
                    vbase + (size_t)(j * 16 + ll) * SEQ + kb * KB + kk * 32 + lh * 16;
                bv.q[0] = *(const uint4*)(vb);
                bv.q[1] = *(const uint4*)(vb + 8);
                acc_o[j] = WMMA_BF16(ap.v, bv.v, acc_o[j]);
            }
        }
    }

    // ---- finalize: O / l, write bf16 into [BL][D] at column h*64+.. ----
    const int b = bh >> 4, h = bh & 15;
#pragma unroll
    for (int j = 0; j < 4; ++j)
#pragma unroll
        for (int r = 0; r < 8; ++r) {
            const int row = qbase + r + 8 * lh;
            const float o = acc_o[j][r] / l_i[r];
            attn[((size_t)(b * SEQ + row)) * D_MODEL + h * 64 + j * 16 + ll] = (bf16_t)o;
        }
}

// ---------------------------------------------------------------------------
// LayerNorm over D=1024 per row; optional f32 and bf16 outputs.
// ---------------------------------------------------------------------------
__global__ __launch_bounds__(256)
void ln_kernel(const float* __restrict__ in, const float* __restrict__ g,
               const float* __restrict__ b, float* __restrict__ outF,
               bf16_t* __restrict__ outB) {
    constexpr int D = D_MODEL;
    __shared__ float red[8];
    const int row = blockIdx.x;
    const float* x = in + (size_t)row * D;

    float xv[4], s = 0.0f;
#pragma unroll
    for (int i = 0; i < 4; ++i) { xv[i] = x[threadIdx.x + i * 256]; s += xv[i]; }
#pragma unroll
    for (int m = 16; m >= 1; m >>= 1) s += __shfl_xor(s, m, 32);
    if ((threadIdx.x & 31) == 0) red[threadIdx.x >> 5] = s;
    __syncthreads();
    float tot = 0.0f;
#pragma unroll
    for (int w = 0; w < 8; ++w) tot += red[w];
    const float mu = tot / D;
    __syncthreads();

    float vs = 0.0f;
#pragma unroll
    for (int i = 0; i < 4; ++i) { float d = xv[i] - mu; vs += d * d; }
#pragma unroll
    for (int m = 16; m >= 1; m >>= 1) vs += __shfl_xor(vs, m, 32);
    if ((threadIdx.x & 31) == 0) red[threadIdx.x >> 5] = vs;
    __syncthreads();
    float vtot = 0.0f;
#pragma unroll
    for (int w = 0; w < 8; ++w) vtot += red[w];
    const float rstd = rsqrtf(vtot / D + 1e-5f);

#pragma unroll
    for (int i = 0; i < 4; ++i) {
        const int c = threadIdx.x + i * 256;
        const float y = (xv[i] - mu) * rstd * g[c] + b[c];
        if (outF) outF[(size_t)row * D + c] = y;
        if (outB) outB[(size_t)row * D + c] = (bf16_t)y;
    }
}

// ---------------------------------------------------------------------------
// Host orchestration
// ---------------------------------------------------------------------------
extern "C" void kernel_launch(void* const* d_in, const int* in_sizes, int n_in,
                              void* d_out, int out_size, void* d_ws, size_t ws_size,
                              hipStream_t stream) {
    (void)in_sizes; (void)n_in; (void)out_size; (void)ws_size;

    const float* x      = (const float*)d_in[0];
    const float* qkv_w  = (const float*)d_in[1];
    const float* qkv_b  = (const float*)d_in[2];
    const float* out_w  = (const float*)d_in[3];
    const float* out_b  = (const float*)d_in[4];
    const float* ln1_g  = (const float*)d_in[5];
    const float* ln1_b  = (const float*)d_in[6];
    const float* ln2_g  = (const float*)d_in[7];
    const float* ln2_b  = (const float*)d_in[8];
    const float* ffn1_w = (const float*)d_in[9];
    const float* ffn1_b = (const float*)d_in[10];
    const float* ffn2_w = (const float*)d_in[11];
    const float* ffn2_b = (const float*)d_in[12];
    float* out = (float*)d_out;

    char* w = (char*)d_ws;
    size_t off = 0;
    auto alloc = [&](size_t bytes) -> void* {
        void* p = w + off;
        off = (off + bytes + 255) & ~(size_t)255;
        return p;
    };

    bf16_t* xb      = (bf16_t*)alloc((size_t)BL * 1024 * 2);
    bf16_t* wqkv_t  = (bf16_t*)alloc((size_t)3072 * 1024 * 2);
    bf16_t* wout_t  = (bf16_t*)alloc((size_t)1024 * 1024 * 2);
    bf16_t* wf1_t   = (bf16_t*)alloc((size_t)4096 * 1024 * 2);
    bf16_t* wf2_t   = (bf16_t*)alloc((size_t)1024 * 4096 * 2);
    bf16_t* qkvb    = (bf16_t*)alloc((size_t)BL * 3072 * 2);
    bf16_t* qhb     = (bf16_t*)alloc((size_t)32 * 2048 * 64 * 2);
    bf16_t* khb     = (bf16_t*)alloc((size_t)32 * 2048 * 64 * 2);
    bf16_t* vtb     = (bf16_t*)alloc((size_t)32 * 64 * 2048 * 2);
    bf16_t* attnb   = (bf16_t*)alloc((size_t)BL * 1024 * 2);
    float*  y1      = (float*)alloc((size_t)BL * 1024 * 4);
    float*  x1f     = (float*)alloc((size_t)BL * 1024 * 4);
    bf16_t* x1b     = (bf16_t*)alloc((size_t)BL * 1024 * 2);
    bf16_t* hb      = (bf16_t*)alloc((size_t)BL * 4096 * 2);
    float*  y2      = (float*)alloc((size_t)BL * 1024 * 4);

    // 1) precision conversions
    cvt_f32_bf16_kernel<<<(BL * 1024) / 256, 256, 0, stream>>>(x, xb, (size_t)BL * 1024);
    transpose_bf16_kernel<<<dim3(3072 / 32, 1024 / 32), 256, 0, stream>>>(qkv_w, wqkv_t, 1024, 3072);
    transpose_bf16_kernel<<<dim3(1024 / 32, 1024 / 32), 256, 0, stream>>>(out_w, wout_t, 1024, 1024);
    transpose_bf16_kernel<<<dim3(4096 / 32, 1024 / 32), 256, 0, stream>>>(ffn1_w, wf1_t, 1024, 4096);
    transpose_bf16_kernel<<<dim3(1024 / 32, 4096 / 32), 256, 0, stream>>>(ffn2_w, wf2_t, 4096, 1024);

    // 2) QKV projection (WMMA) -> bf16
    gemm_bf16_kernel<3><<<dim3(3072 / 128, BL / 128), 256, 0, stream>>>(
        xb, wqkv_t, qkv_b, nullptr, nullptr, qkvb, BL, 3072, 1024);

    // 3) RoPE + head scatter (+1/sqrt(hd) into Q, V transposed)
    rope_scatter_kernel<<<(2 * 2048 * 16 * 32) / 256, 256, 0, stream>>>(qkvb, qhb, khb, vtb);

    // 4) flash attention (WMMA)
    attn_kernel<<<dim3(SEQ / 128, BATCH * NHEAD), 256, 0, stream>>>(qhb, khb, vtb, attnb);

    // 5) output projection + residual (WMMA)
    gemm_bf16_kernel<1><<<dim3(1024 / 128, BL / 128), 256, 0, stream>>>(
        attnb, wout_t, out_b, x, y1, nullptr, BL, 1024, 1024);

    // 6) LN1 -> f32 residual copy + bf16 activations
    ln_kernel<<<BL, 256, 0, stream>>>(y1, ln1_g, ln1_b, x1f, x1b);

    // 7) FFN1 + exact GELU (WMMA)
    gemm_bf16_kernel<2><<<dim3(4096 / 128, BL / 128), 256, 0, stream>>>(
        x1b, wf1_t, ffn1_b, nullptr, nullptr, hb, BL, 4096, 1024);

    // 8) FFN2 + residual (WMMA)
    gemm_bf16_kernel<1><<<dim3(1024 / 128, BL / 128), 256, 0, stream>>>(
        hb, wf2_t, ffn2_b, x1f, y2, nullptr, BL, 1024, 4096);

    // 9) LN2 -> final f32 output
    ln_kernel<<<BL, 256, 0, stream>>>(y2, ln2_g, ln2_b, out, nullptr);
}
